// VAErecV2_36369783063089
// MI455X (gfx1250) — matrix operand order repaired and checked by reading, not compiled, not run on hardware
//
#include <hip/hip_runtime.h>
#include <hip/hip_bf16.h>

// ---------------- problem constants ----------------
#define NX   144
#define NXR  128
#define NZC  128
#define TT   90
#define BB   2048
#define NH0  300
#define NH1  200

// padded K dims (multiples of 32)
#define KP_X   160   // 144 -> 160
#define KP_H   128   // 128
#define KP_H0  320   // 300 -> 320
#define KP_H1  224   // 200 -> 224

typedef unsigned short u16;
typedef unsigned int   u32;
typedef __attribute__((ext_vector_type(16))) __bf16         v16bf;
typedef __attribute__((ext_vector_type(16))) unsigned short v16u;
typedef __attribute__((ext_vector_type(8)))  float          v8f;

__device__ __forceinline__ u16 f2bf(float f) {
  u32 u = __builtin_bit_cast(u32, f);
  u32 r = (u + 0x7FFFu + ((u >> 16) & 1u)) >> 16;
  return (u16)r;
}

// ---------------- weight pack: f32 [K,N] -> bf16 packed [(Kpad/32)][Npad][32] ----------------
// layout: dst[(kb*Npad + n)*32 + i] = bf16(W[(kb*32+i)*N + n]) or 0 (zero-pad K and N)
__global__ void pack_w_kernel(const float* __restrict__ W, int K, int N,
                              int Kpad, int Npad, u16* __restrict__ dst) {
  int idx = blockIdx.x * blockDim.x + threadIdx.x;
  int total = Kpad * Npad;
  if (idx >= total) return;
  int i    = idx & 31;
  int rest = idx >> 5;
  int n    = rest % Npad;
  int kb   = rest / Npad;
  int k    = kb * 32 + i;
  float v  = (k < K && n < N) ? W[(size_t)k * N + n] : 0.0f;
  dst[idx] = f2bf(v);
}

// ---------------- f32 [M,K] -> bf16 [M,Kpad] zero-padded ----------------
__global__ void cvt_pad_kernel(const float* __restrict__ src, int K, int Kpad,
                               int M, u16* __restrict__ dst) {
  int idx = blockIdx.x * blockDim.x + threadIdx.x;
  if (idx >= M * Kpad) return;
  int k = idx % Kpad;
  int m = idx / Kpad;
  dst[idx] = (k < K) ? f2bf(src[(size_t)m * K + k]) : (u16)0;
}

// ---------------- zero init (f32 + bf16 state) ----------------
__global__ void zero2_kernel(float* __restrict__ f, int nf, u16* __restrict__ b, int nb) {
  int idx = blockIdx.x * blockDim.x + threadIdx.x;
  if (idx < nf) f[idx] = 0.0f;
  if (idx < nb) b[idx] = 0;
}

// ---------------- generic WMMA bf16 GEMM, NT N-tiles per wave sharing one A fragment ----------
// C[M,Npad] = act( A[M,Kpad](bf16) @ Wp + bias[n<Nvalid] + addend[m,n] )
// per K-chunk: 1 A-fragment load, NT B-fragment loads, NT v_wmma_f32_16x16x32_bf16.
// act: 0=none 1=tanh 2=relu.  Cb: bf16 out (full Npad cols; pads are exact 0 because
// padded weight cols/bias are 0).  Cf: f32 out, only cols n < Nvalid.
template <int NT>
__global__ void gemm_bf16_kernel(const u16* __restrict__ A, int lda,
                                 const u16* __restrict__ Wp, int Npad,
                                 const float* __restrict__ bias,
                                 const float* __restrict__ addend, int ldadd,
                                 u16* __restrict__ Cb, int ldcb,
                                 float* __restrict__ Cf, int ldcf,
                                 int M, int Kpad, int Nvalid, int act) {
  int wave = blockIdx.x * (blockDim.x >> 5) + (threadIdx.x >> 5);
  int lane = threadIdx.x & 31;
  int ntn  = Npad >> 4;
  int ngrp = ntn / NT;
  int tiles = (M >> 4) * ngrp;
  if (wave >= tiles) return;            // wave-uniform: EXEC stays all-ones for WMMA
  int tm = wave / ngrp, tg = wave % ngrp;
  int m0 = tm << 4, n0 = (tg * NT) << 4;
  int col  = lane & 15;                 // A row offset, W column offset, C column offset
  int half = lane >> 4;

  const u16* arow  = A  + (size_t)(m0 + col) * lda;
  const u16* wbase = Wp + (size_t)(n0 + col) * 32 + half * 16;

  v8f c[NT];
#pragma unroll
  for (int t = 0; t < NT; ++t) c[t] = (v8f){0.f, 0.f, 0.f, 0.f, 0.f, 0.f, 0.f, 0.f};

  int nkb = Kpad >> 5;
  for (int kb = 0; kb < nkb; ++kb) {
    v16u au;
    int ka = kb * 32 + 16 * half;       // A lane K set: 16h..16h+15 (contig)
#pragma unroll
    for (int j = 0; j < 8; ++j) {
      au[j]     = arow[ka + j];
      au[8 + j] = arow[ka + 8 + j];
    }
    const u16* wp = wbase + (size_t)kb * Npad * 32;  // B lane K set: 16h..16h+15 (contig)
    v16u bu[NT];
#pragma unroll
    for (int t = 0; t < NT; ++t) {
#pragma unroll
      for (int j = 0; j < 16; ++j) bu[t][j] = wp[t * 512 + j];   // next N-tile: 16 cols * 32
    }
    v16bf a = __builtin_bit_cast(v16bf, au);
#pragma unroll
    for (int t = 0; t < NT; ++t) {
      c[t] = __builtin_amdgcn_wmma_f32_16x16x32_bf16(
                 false, a, false, __builtin_bit_cast(v16bf, bu[t]),
                 (short)0, c[t], false, false);
    }
  }

#pragma unroll
  for (int t = 0; t < NT; ++t) {
    int n = n0 + t * 16 + col;
    float bv = 0.0f;
    if (bias != nullptr && n < Nvalid) bv = bias[n];
#pragma unroll
    for (int r = 0; r < 8; ++r) {
      int m = m0 + 8 * half + r;        // C layout: VGPR r -> row r + 8*half, col = lane&15
      float v = c[t][r] + bv;
      if (addend != nullptr) v += addend[(size_t)m * ldadd + n];
      if (act == 1)      v = tanhf(v);
      else if (act == 2) v = fmaxf(v, 0.0f);
      if (Cb != nullptr)               Cb[(size_t)m * ldcb + n] = f2bf(v);
      if (Cf != nullptr && n < Nvalid) Cf[(size_t)m * ldcf + n] = v;
    }
  }
}

// ---------------- GRU elementwise update (PyTorch gate order r,z,n) ----------------
__global__ void gru_elem_kernel(const float* __restrict__ gi, const float* __restrict__ gh,
                                const float* __restrict__ hprev,
                                float* __restrict__ hf, u16* __restrict__ hb) {
  int idx = blockIdx.x * blockDim.x + threadIdx.x;
  if (idx >= BB * NXR) return;
  int b = idx / NXR, j = idx % NXR;
  const float* gib = gi + (size_t)b * 3 * NXR;
  const float* ghb = gh + (size_t)b * 3 * NXR;
  float r = 1.0f / (1.0f + __expf(-(gib[j]           + ghb[j])));
  float z = 1.0f / (1.0f + __expf(-(gib[NXR + j]     + ghb[NXR + j])));
  float n = tanhf(gib[2 * NXR + j] + r * ghb[2 * NXR + j]);
  float h = (1.0f - z) * n + z * hprev[idx];
  hf[idx] = h;
  hb[idx] = f2bf(h);
}

// ================= host side =================
static inline size_t alignUp256(size_t x) { return (x + 255) & ~(size_t)255; }

struct Bump {
  char* p; size_t used;
  void* take(size_t bytes) { void* r = p + used; used = alignUp256(used + bytes); return r; }
};

static void launch_pack(hipStream_t s, const float* W, int K, int N, int Kpad, int Npad, u16* dst) {
  int total = Kpad * Npad;
  pack_w_kernel<<<(total + 255) / 256, 256, 0, s>>>(W, K, N, Kpad, Npad, dst);
}

static void launch_gemm(hipStream_t s, const u16* A, int lda, const u16* Wp, int Npad,
                        const float* bias, const float* addend, int ldadd,
                        u16* Cb, int ldcb, float* Cf, int ldcf,
                        int M, int Kpad, int Nvalid, int act) {
  int ntn = Npad / 16;
  if ((ntn & 3) == 0) {
    int tiles = (M / 16) * (ntn / 4);
    int blocks = (tiles + 7) / 8;       // 8 waves (256 thr) per block, wave32
    gemm_bf16_kernel<4><<<blocks, 256, 0, s>>>(A, lda, Wp, Npad, bias, addend, ldadd,
                                               Cb, ldcb, Cf, ldcf, M, Kpad, Nvalid, act);
  } else if ((ntn & 1) == 0) {
    int tiles = (M / 16) * (ntn / 2);
    int blocks = (tiles + 7) / 8;
    gemm_bf16_kernel<2><<<blocks, 256, 0, s>>>(A, lda, Wp, Npad, bias, addend, ldadd,
                                               Cb, ldcb, Cf, ldcf, M, Kpad, Nvalid, act);
  } else {
    int tiles = (M / 16) * ntn;
    int blocks = (tiles + 7) / 8;
    gemm_bf16_kernel<1><<<blocks, 256, 0, s>>>(A, lda, Wp, Npad, bias, addend, ldadd,
                                               Cb, ldcb, Cf, ldcf, M, Kpad, Nvalid, act);
  }
}

extern "C" void kernel_launch(void* const* d_in, const int* in_sizes, int n_in,
                              void* d_out, int out_size, void* d_ws, size_t ws_size,
                              hipStream_t stream) {
  (void)in_sizes; (void)n_in; (void)out_size; (void)ws_size;
  const float* x     = (const float*)d_in[0];
  const float* e_Wi  = (const float*)d_in[1];
  const float* e_Wh  = (const float*)d_in[2];
  const float* e_bi  = (const float*)d_in[3];
  const float* e_bh  = (const float*)d_in[4];
  const float* em_W1 = (const float*)d_in[5];
  const float* em_b1 = (const float*)d_in[6];
  const float* em_W2 = (const float*)d_in[7];
  const float* em_b2 = (const float*)d_in[8];
  const float* mu_W  = (const float*)d_in[9];
  const float* mu_b  = (const float*)d_in[10];
  const float* lv_W  = (const float*)d_in[11];
  const float* lv_b  = (const float*)d_in[12];
  const float* ip_W1 = (const float*)d_in[13];
  const float* ip_b1 = (const float*)d_in[14];
  const float* ip_W2 = (const float*)d_in[15];
  const float* ip_b2 = (const float*)d_in[16];
  const float* ipo_W = (const float*)d_in[17];
  const float* ipo_b = (const float*)d_in[18];
  const float* d_Wi  = (const float*)d_in[19];
  const float* d_Wh  = (const float*)d_in[20];
  const float* d_bi  = (const float*)d_in[21];
  const float* d_bh  = (const float*)d_in[22];
  const float* dm_W1 = (const float*)d_in[23];
  const float* dm_b1 = (const float*)d_in[24];
  const float* dm_W2 = (const float*)d_in[25];
  const float* dm_b2 = (const float*)d_in[26];
  const float* do_W  = (const float*)d_in[27];
  const float* do_b  = (const float*)d_in[28];

  float* out    = (float*)d_out;
  float* out_mu = out + (size_t)TT * BB * NX;
  float* out_lv = out_mu + (size_t)BB * NZC;

  Bump ws{(char*)d_ws, 0};
  // packed weights (bf16)
  u16* p_eWi  = (u16*)ws.take((size_t)KP_X  * 384 * 2);
  u16* p_eWh  = (u16*)ws.take((size_t)KP_H  * 384 * 2);
  u16* p_emW1 = (u16*)ws.take((size_t)KP_H  * KP_H0 * 2);
  u16* p_emW2 = (u16*)ws.take((size_t)KP_H0 * KP_H1 * 2);
  u16* p_muW  = (u16*)ws.take((size_t)KP_H1 * 128 * 2);
  u16* p_lvW  = (u16*)ws.take((size_t)KP_H1 * 128 * 2);
  u16* p_ipW1 = (u16*)ws.take((size_t)KP_H  * KP_H0 * 2);
  u16* p_ipW2 = (u16*)ws.take((size_t)KP_H0 * KP_H1 * 2);
  u16* p_ipoW = (u16*)ws.take((size_t)KP_H1 * KP_X * 2);
  u16* p_dWix = (u16*)ws.take((size_t)KP_X  * 384 * 2);
  u16* p_dWiz = (u16*)ws.take((size_t)KP_H  * 384 * 2);
  u16* p_dWh  = (u16*)ws.take((size_t)KP_H  * 384 * 2);
  u16* p_dmW1 = (u16*)ws.take((size_t)KP_H  * KP_H0 * 2);
  u16* p_dmW2 = (u16*)ws.take((size_t)KP_H0 * KP_H1 * 2);
  u16* p_doW  = (u16*)ws.take((size_t)KP_H1 * KP_X * 2);
  // activations
  u16*   xb   = (u16*)ws.take((size_t)BB * KP_X * 2);
  float* h_f  = (float*)ws.take((size_t)BB * NXR * 4);
  u16*   h_b  = (u16*)ws.take((size_t)BB * NXR * 2);
  float* gi   = (float*)ws.take((size_t)BB * 384 * 4);
  float* gh   = (float*)ws.take((size_t)BB * 384 * 4);
  u16*   hm1  = (u16*)ws.take((size_t)BB * KP_H0 * 2);
  u16*   hm2  = (u16*)ws.take((size_t)BB * KP_H1 * 2);
  u16*   z_b  = (u16*)ws.take((size_t)BB * 128 * 2);
  u16*   xp   = (u16*)ws.take((size_t)BB * KP_X * 2);
  float* giz  = (float*)ws.take((size_t)BB * 384 * 4);
  float* hd_f = (float*)ws.take((size_t)BB * NXR * 4);
  u16*   hd_b = (u16*)ws.take((size_t)BB * NXR * 2);

  // ---- pack all weights (Npad chosen = consumer's Kpad so pad cols are exact zeros) ----
  launch_pack(stream, e_Wi,  NX,       3 * NXR, KP_X,  384,   p_eWi);
  launch_pack(stream, e_Wh,  NXR,      3 * NXR, KP_H,  384,   p_eWh);
  launch_pack(stream, em_W1, NXR,      NH0,     KP_H,  KP_H0, p_emW1);
  launch_pack(stream, em_W2, NH0,      NH1,     KP_H0, KP_H1, p_emW2);
  launch_pack(stream, mu_W,  NH1,      NZC,     KP_H1, 128,   p_muW);
  launch_pack(stream, lv_W,  NH1,      NZC,     KP_H1, 128,   p_lvW);
  launch_pack(stream, ip_W1, NZC,      NH0,     KP_H,  KP_H0, p_ipW1);
  launch_pack(stream, ip_W2, NH0,      NH1,     KP_H0, KP_H1, p_ipW2);
  launch_pack(stream, ipo_W, NH1,      NX,      KP_H1, KP_X,  p_ipoW);
  launch_pack(stream, d_Wi,              NX,  3 * NXR, KP_X, 384, p_dWix); // rows 0..143 (x part)
  launch_pack(stream, d_Wi + (size_t)NX * 3 * NXR, NXR, 3 * NXR, KP_H, 384, p_dWiz); // rows 144..271 (z part)
  launch_pack(stream, d_Wh,  NXR,      3 * NXR, KP_H,  384,   p_dWh);
  launch_pack(stream, dm_W1, NXR,      NH0,     KP_H,  KP_H0, p_dmW1);
  launch_pack(stream, dm_W2, NH0,      NH1,     KP_H0, KP_H1, p_dmW2);
  launch_pack(stream, do_W,  NH1,      NX,      KP_H1, KP_X,  p_doW);

  // ---- encoder GRU over time ----
  {
    int n = BB * NXR;
    zero2_kernel<<<(n + 255) / 256, 256, 0, stream>>>(h_f, n, h_b, n);
  }
  for (int t = 0; t < TT; ++t) {
    const float* xt = x + (size_t)t * BB * NX;
    int nc = BB * KP_X;
    cvt_pad_kernel<<<(nc + 255) / 256, 256, 0, stream>>>(xt, NX, KP_X, BB, xb);
    // gi = x_t @ e_Wi + e_bi   ;   gh = h @ e_Wh + e_bh
    launch_gemm(stream, xb,  KP_X, p_eWi, 384, e_bi, nullptr, 0, nullptr, 0, gi, 384, BB, KP_X, 384, 0);
    launch_gemm(stream, h_b, KP_H, p_eWh, 384, e_bh, nullptr, 0, nullptr, 0, gh, 384, BB, KP_H, 384, 0);
    int ng = BB * NXR;
    gru_elem_kernel<<<(ng + 255) / 256, 256, 0, stream>>>(gi, gh, h_f, h_f, h_b);
  }

  // ---- encoder head: mu / logvar (z = mu) ----
  launch_gemm(stream, h_b, KP_H,  p_emW1, KP_H0, em_b1, nullptr, 0, hm1, KP_H0, nullptr, 0, BB, KP_H,  NH0, 1);
  launch_gemm(stream, hm1, KP_H0, p_emW2, KP_H1, em_b2, nullptr, 0, hm2, KP_H1, nullptr, 0, BB, KP_H0, NH1, 1);
  launch_gemm(stream, hm2, KP_H1, p_muW,  128,   mu_b,  nullptr, 0, z_b, 128,  out_mu, 128, BB, KP_H1, 128, 0);
  launch_gemm(stream, hm2, KP_H1, p_lvW,  128,   lv_b,  nullptr, 0, nullptr, 0, out_lv, 128, BB, KP_H1, 128, 0);

  // ---- init pose from z ----
  launch_gemm(stream, z_b, KP_H,  p_ipW1, KP_H0, ip_b1, nullptr, 0, hm1, KP_H0, nullptr, 0, BB, KP_H,  NH0, 2);
  launch_gemm(stream, hm1, KP_H0, p_ipW2, KP_H1, ip_b2, nullptr, 0, hm2, KP_H1, nullptr, 0, BB, KP_H0, NH1, 2);
  launch_gemm(stream, hm2, KP_H1, p_ipoW, KP_X,  ipo_b, nullptr, 0, xp,  KP_X,  nullptr, 0, BB, KP_H1, NX, 0);

  // ---- precompute z contribution to decoder input gates: giz = z @ d_Wi[144:,:] + d_bi ----
  launch_gemm(stream, z_b, KP_H, p_dWiz, 384, d_bi, nullptr, 0, nullptr, 0, giz, 384, BB, KP_H, 384, 0);

  // ---- decoder autoregressive loop ----
  {
    int n = BB * NXR;
    zero2_kernel<<<(n + 255) / 256, 256, 0, stream>>>(hd_f, n, hd_b, n);
  }
  for (int t = 0; t < TT; ++t) {
    // gi = xp @ d_Wi[:144,:] + giz   ;   gh = hd @ d_Wh + d_bh
    launch_gemm(stream, xp,   KP_X, p_dWix, 384, nullptr, giz, 384, nullptr, 0, gi, 384, BB, KP_X, 384, 0);
    launch_gemm(stream, hd_b, KP_H, p_dWh,  384, d_bh, nullptr, 0,  nullptr, 0, gh, 384, BB, KP_H, 384, 0);
    int ng = BB * NXR;
    gru_elem_kernel<<<(ng + 255) / 256, 256, 0, stream>>>(gi, gh, hd_f, hd_f, hd_b);
    // decoder MLP -> x_i (write f32 to d_out[t] and bf16 to xp for next step)
    launch_gemm(stream, hd_b, KP_H,  p_dmW1, KP_H0, dm_b1, nullptr, 0, hm1, KP_H0, nullptr, 0, BB, KP_H,  NH0, 1);
    launch_gemm(stream, hm1,  KP_H0, p_dmW2, KP_H1, dm_b2, nullptr, 0, hm2, KP_H1, nullptr, 0, BB, KP_H0, NH1, 1);
    float* xrec_t = out + (size_t)t * BB * NX;
    launch_gemm(stream, hm2,  KP_H1, p_doW,  KP_X,  do_b,  nullptr, 0, xp, KP_X, xrec_t, NX, BB, KP_H1, NX, 0);
  }
}
